// ReservoirBlock_62569083568242
// MI455X (gfx1250) — matrix operand order repaired and checked by reading, not compiled, not run on hardware
//
#include <hip/hip_runtime.h>
#include <hip/hip_bf16.h>
#include <math.h>

// ---------------- problem constants ----------------
#define BATCH_   4
#define SEQ_     2048
#define HID_     1024
#define RES_     2048
#define LEAK_    0.3f
#define RB_NBLK  32          // persistent blocks for the recurrence (grid barrier)

typedef __bf16 bf16_t;
typedef bf16_t v16bf __attribute__((ext_vector_type(16)));
typedef bf16_t v8bf  __attribute__((ext_vector_type(8)));
typedef float  v8f   __attribute__((ext_vector_type(8)));

__device__ __forceinline__ v8bf ldv8(const bf16_t* p) { return *(const v8bf*)p; }

__device__ __forceinline__ v16bf cat16(v8bf lo, v8bf hi) {
    v16bf r;
#pragma unroll
    for (int i = 0; i < 8; ++i) { r[i] = lo[i]; r[i + 8] = hi[i]; }
    return r;
}

// A fragment (16x32 bf16, row-major source, lane row = lane&15):
//   K offsets: khalf*8 + [0..7]  and  khalf*8 + 16 + [0..7]
__device__ __forceinline__ v16bf load_fragA(const bf16_t* rowp, int k0, int lane) {
    const int kh = (lane >> 4) << 3;               // 0 or 8
    return cat16(ldv8(rowp + k0 + kh), ldv8(rowp + k0 + kh + 16));
}
// B fragment from B^T (N-major [N][K], lane col = lane&15):
//   K offsets: khalf*8 + [0..7]  and  khalf*8 + 16 + [0..7]
__device__ __forceinline__ v16bf load_fragB(const bf16_t* rowp, int k0, int lane) {
    const int kh = (lane >> 4) << 3;               // 0 or 8
    return cat16(ldv8(rowp + k0 + kh), ldv8(rowp + k0 + kh + 16));
}

// ---------------- prep kernels ----------------
__global__ void k_f32_to_bf16(const float* __restrict__ src, bf16_t* __restrict__ dst, int n) {
    int i = blockIdx.x * blockDim.x + threadIdx.x;
    if (i < n) dst[i] = (bf16_t)src[i];
}

// src [K][N] f32  ->  dst [N][K] bf16   (coalesced writes)
__global__ void k_transpose_bf16(const float* __restrict__ src, bf16_t* __restrict__ dst,
                                 int K, int N) {
    int i = blockIdx.x * blockDim.x + threadIdx.x;
    if (i < K * N) {
        int nn = i / K, kk = i - nn * K;
        dst[i] = (bf16_t)src[(size_t)kk * N + nn];
    }
}

__global__ void k_zero32(unsigned int* __restrict__ p, int n) {
    int i = blockIdx.x * blockDim.x + threadIdx.x;
    if (i < n) p[i] = 0u;
}

// ---------------- generic bf16 WMMA GEMM ----------------
// C[M,N] = A[M,K](bf16 row-major) @ B^T[N,K](bf16) + bias
// one wave -> 16 rows x 64 cols; MODE 0: bf16 row-major out
//                                MODE 1: fp32 scatter to Z[s][b][RES] (row m = b*SEQ+s)
//                                MODE 2: fp32 row-major out
template <int MODE>
__global__ __launch_bounds__(256) void gemm_bf16_wmma(
    const bf16_t* __restrict__ A, const bf16_t* __restrict__ BT,
    const float* __restrict__ bias, void* __restrict__ Out,
    int M, int N, int K)
{
    const int lane = threadIdx.x & 31;
    const int wave = (blockIdx.x * blockDim.x + threadIdx.x) >> 5;
    const int Mt = M >> 4, Ng = N >> 6;
    if (wave >= Mt * Ng) return;                       // wave-uniform branch
    const int mt = wave % Mt, ng = wave / Mt;
    const int col = lane & 15;
    const bf16_t* arow = A + (size_t)((mt << 4) + col) * K;   // lane row = lane&15
    const bf16_t* brow[4];
#pragma unroll
    for (int j = 0; j < 4; ++j)
        brow[j] = BT + (size_t)((ng << 6) + (j << 4) + col) * K;

    v8f acc[4] = {};
    for (int k0 = 0; k0 < K; k0 += 32) {
        const v16bf a = load_fragA(arow, k0, lane);
#pragma unroll
        for (int j = 0; j < 4; ++j) {
            const v16bf b = load_fragB(brow[j], k0, lane);
            acc[j] = __builtin_amdgcn_wmma_f32_16x16x32_bf16(
                         false, a, false, b, (short)0, acc[j], false, false);
        }
    }
    // epilogue: C element (lane,v) -> row = v + 8*(lane>>4), col = lane&15
    const int rbase = (mt << 4) + ((lane >> 4) << 3);
#pragma unroll
    for (int j = 0; j < 4; ++j) {
        const int n = (ng << 6) + (j << 4) + col;
        const float bv = bias ? bias[n] : 0.0f;
#pragma unroll
        for (int v = 0; v < 8; ++v) {
            const int r = rbase + v;
            const float val = acc[j][v] + bv;
            if (MODE == 0) {
                ((bf16_t*)Out)[(size_t)r * N + n] = (bf16_t)val;
            } else if (MODE == 1) {                    // Z[s][b][RES], r = b*SEQ+s
                const int b = r >> 11, s = r & (SEQ_ - 1);
                ((float*)Out)[(((size_t)s * BATCH_ + b) << 11) + n] = val;
            } else {
                ((float*)Out)[(size_t)r * N + n] = val;
            }
        }
    }
}

// ---------------- persistent recurrence kernel ----------------
// 32 blocks x 256 thr; block owns 64 output columns of the reservoir.
// Per step: [16(pad) x 2048] bf16 state  @  W_res^T slice  via WMMA,
// fp32 leaky tanh update, device-wide barrier.
__global__ __launch_bounds__(256, 1) void reservoir_recur_wmma(
    const float*  __restrict__ Z,          // [SEQ][4][RES] fp32 pre-computed input drive
    const bf16_t* __restrict__ WresT,      // [RES][RES] bf16 (N-major)
    float*        __restrict__ state_f32,  // [4][RES]
    bf16_t*       __restrict__ state_bf,   // [2][16][RES] double buffered, rows 4..15 = 0
    bf16_t*       __restrict__ states_out, // [4*SEQ][RES] bf16 (row = b*SEQ+s)
    int*          __restrict__ cnt)        // [SEQ] arrival counters (pre-zeroed)
{
    __shared__ float red[4][8][32];
    const int lane  = threadIdx.x & 31;
    const int wave  = threadIdx.x >> 5;
    const int ntile = wave & 3;                 // which 16-col tile of our 64
    const int khalf = wave >> 2;                // K split: 0..1023 / 1024..2047
    const int col   = lane & 15;
    const int n     = blockIdx.x * 64 + ntile * 16 + col;
    const bf16_t* brow = WresT + (size_t)n * RES_ + khalf * 1024;
    const int arow = lane & 15;                 // padded batch row
    const int aoff = khalf * 1024;

    for (int t = 0; t < SEQ_; ++t) {
        const bf16_t* abase = state_bf + (size_t)(t & 1) * 16 * RES_
                                       + (size_t)arow * RES_ + aoff;
        v8f acc = {};
#pragma unroll 2
        for (int kk = 0; kk < 1024; kk += 32) {
            const v16bf a = load_fragA(abase, kk, lane);
            const v16bf b = load_fragB(brow,  kk, lane);
            acc = __builtin_amdgcn_wmma_f32_16x16x32_bf16(
                      false, a, false, b, (short)0, acc, false, false);
        }
        if (khalf == 1) {
#pragma unroll
            for (int v = 0; v < 8; ++v) red[ntile][v][lane] = acc[v];
        }
        __syncthreads();
        if (khalf == 0) {
#pragma unroll
            for (int v = 0; v < 8; ++v) acc[v] += red[ntile][v][lane];
            if (lane < 16) {                    // rows 0..3 are the real batches
#pragma unroll
                for (int v = 0; v < BATCH_; ++v) {
                    const float pre = Z[(((size_t)t * BATCH_ + v) << 11) + n] + acc[v];
                    const float so  = state_f32[(v << 11) + n];
                    const float sn  = (1.0f - LEAK_) * so + LEAK_ * tanhf(pre);
                    state_f32[(v << 11) + n] = sn;
                    state_bf[(size_t)((t + 1) & 1) * 16 * RES_ + (v << 11) + n] = (bf16_t)sn;
                    states_out[(((size_t)v * SEQ_ + t) << 11) + n] = (bf16_t)sn;
                }
            }
        }
        // -------- device-wide step barrier --------
        __builtin_amdgcn_fence(__ATOMIC_RELEASE, "agent");
        __syncthreads();
        if (threadIdx.x == 0) {
            __hip_atomic_fetch_add(&cnt[t], 1, __ATOMIC_RELAXED, __HIP_MEMORY_SCOPE_AGENT);
            while (__hip_atomic_load(&cnt[t], __ATOMIC_RELAXED, __HIP_MEMORY_SCOPE_AGENT)
                   < RB_NBLK)
                __builtin_amdgcn_s_sleep(1);
        }
        __syncthreads();
        __builtin_amdgcn_fence(__ATOMIC_ACQUIRE, "agent");
    }
}

// ---------------- launcher ----------------
extern "C" void kernel_launch(void* const* d_in, const int* in_sizes, int n_in,
                              void* d_out, int out_size, void* d_ws, size_t ws_size,
                              hipStream_t stream) {
    const float* hs      = (const float*)d_in[0];   // [4,2048,1024]
    const float* W_write = (const float*)d_in[1];   // [1024,1024]
    const float* b_write = (const float*)d_in[2];   // [1024]
    const float* W_in    = (const float*)d_in[3];   // [1024,2048]
    const float* W_res   = (const float*)d_in[4];   // [2048,2048]
    const float* W_read  = (const float*)d_in[5];   // [2048,1024]
    const float* b_read  = (const float*)d_in[6];   // [1024]
    float* out = (float*)d_out;

    char* ws = (char*)d_ws;
    constexpr size_t MB = 1024ull * 1024ull;
    bf16_t* hs_bf   = (bf16_t*)(ws);                 // 16 MB  [8192][1024]
    bf16_t* u_bf    = (bf16_t*)(ws + 16 * MB);       // 16 MB  [8192][1024]
    bf16_t* st_bf   = (bf16_t*)(ws);                 // 32 MB  [8192][2048] (reuses hs/u)
    float*  Zbuf    = (float*) (ws + 32 * MB);       // 64 MB  [2048][4][2048]
    bf16_t* WWT     = (bf16_t*)(ws + 96 * MB);       //  2 MB  W_write^T [1024][1024]
    bf16_t* WIT     = (bf16_t*)(ws + 98 * MB);       //  4 MB  W_in^T   [2048][1024]
    bf16_t* WRT     = (bf16_t*)(ws + 102 * MB);      //  8 MB  W_res^T  [2048][2048]
    bf16_t* WOT     = (bf16_t*)(ws + 110 * MB);      //  4 MB  W_read^T [1024][2048]
    float*  sf32    = (float*) (ws + 114 * MB);      // 32 KB  state fp32 [4][2048]
    bf16_t* sbf     = (bf16_t*)(ws + 114 * MB + 32768);      // 128 KB [2][16][2048]
    int*    cnt     = (int*)   (ws + 114 * MB + 32768 + 131072); // 8 KB

    const int T = 256;
    // zero state + barrier counters (contiguous region), every launch
    {
        const int nz = (32768 + 131072 + SEQ_ * 4) / 4;
        k_zero32<<<(nz + T - 1) / T, T, 0, stream>>>((unsigned int*)sf32, nz);
    }
    // bf16 conversions / weight transposes
    k_f32_to_bf16<<<(BATCH_ * SEQ_ * HID_ + T - 1) / T, T, 0, stream>>>(hs, hs_bf, BATCH_ * SEQ_ * HID_);
    k_transpose_bf16<<<(HID_ * HID_ + T - 1) / T, T, 0, stream>>>(W_write, WWT, HID_, HID_);
    k_transpose_bf16<<<(HID_ * RES_ + T - 1) / T, T, 0, stream>>>(W_in,   WIT, HID_, RES_);
    k_transpose_bf16<<<(RES_ * RES_ + T - 1) / T, T, 0, stream>>>(W_res,  WRT, RES_, RES_);
    k_transpose_bf16<<<(RES_ * HID_ + T - 1) / T, T, 0, stream>>>(W_read, WOT, RES_, HID_);

    const int Mrows = BATCH_ * SEQ_;                          // 8192
    // u = hs @ W_write + b_write            (bf16 out)
    gemm_bf16_wmma<0><<<(Mrows / 16) * (HID_ / 64) / 8, T, 0, stream>>>(
        hs_bf, WWT, b_write, u_bf, Mrows, HID_, HID_);
    // Z = u @ W_in                          (fp32, scattered to [s][b][RES])
    gemm_bf16_wmma<1><<<(Mrows / 16) * (RES_ / 64) / 8, T, 0, stream>>>(
        u_bf, WIT, nullptr, Zbuf, Mrows, RES_, HID_);
    // sequential leaky-tanh recurrence (persistent, grid-barrier per step)
    reservoir_recur_wmma<<<RB_NBLK, T, 0, stream>>>(Zbuf, WRT, sf32, sbf, st_bf, cnt);
    // out = states @ W_read + b_read        (fp32 out)
    gemm_bf16_wmma<2><<<(Mrows / 16) * (HID_ / 64) / 8, T, 0, stream>>>(
        st_bf, WOT, b_read, out, Mrows, HID_, RES_);
}
